// Attention_7404523618850
// MI455X (gfx1250) — matrix operand order repaired and checked by reading, not compiled, not run on hardware
//
#include <hip/hip_runtime.h>
#include <stdint.h>

#define B_ 32
#define S_ 1024
#define H_ 512
#define E_ 1024            // 2*H
#define NEG_ (-10000000000.0f)
#define ASTRIDE 516        // u32 stride per A row in LDS (512 pairs + pad, keeps 16B align)

typedef float  v8f   __attribute__((ext_vector_type(8)));
typedef __bf16 v16bf __attribute__((ext_vector_type(16)));

union BFrag {
  uint4 q[2];
  v16bf v;
};

__device__ __forceinline__ uint32_t packbf2(float lo, float hi) {
  // round-to-nearest-even fp32 -> bf16, pack (hi<<16)|lo
  uint32_t x = __float_as_uint(lo);
  uint32_t y = __float_as_uint(hi);
  x = (x + 0x7FFFu + ((x >> 16) & 1u)) >> 16;
  y = (y + 0x7FFFu + ((y >> 16) & 1u)) >> 16;
  return (y << 16) | (x & 0xFFFFu);
}

// ---------------------------------------------------------------------------
// Kernel 1: convert W_e (rows 512..1535 of W_attn, K=1024 x N=512, row-major)
// to bf16 packed directly in per-lane WMMA B-fragment order.
// B 32x16 (bf16) k order used here (contiguous, same as A): lane = half*16 + n, half=0 -> K=0..15, half=1 -> K=16..31,
// VGPR j holds K = koff + 2j (low 16b) and koff + 2j + 1 (high 16b).
// Storage: [kt][nt][chunk(2)][lane(32)][4 u32]  -> two coalesced b128 loads/frag.
// ---------------------------------------------------------------------------
__global__ __launch_bounds__(256) void pack_we_kernel(const float* __restrict__ Wattn,
                                                      uint32_t* __restrict__ wpack) {
  int t    = blockIdx.x * 256 + threadIdx.x;  // 0 .. 262143
  int j    = t & 3;
  int lane = (t >> 2) & 31;
  int j4   = (t >> 7) & 1;
  int nt   = (t >> 8) & 31;
  int kt   = t >> 13;
  int jj   = j4 * 4 + j;                      // B VGPR index 0..7
  int n    = nt * 16 + (lane & 15);
  int koff = (lane & 16) ? 16 : 0;
  int k    = kt * 32 + koff + 2 * jj;
  const float* We = Wattn + (size_t)H_ * H_;  // W_attn rows [512,1536)
  float lo = We[(size_t)k * H_ + n];
  float hi = We[(size_t)(k + 1) * H_ + n];
  wpack[t] = packbf2(lo, hi);
}

// ---------------------------------------------------------------------------
// Kernel 2: proj[b][h] = hidden[b] . W_h[:,h] + b_attn[h]   (tiny)
// ---------------------------------------------------------------------------
__global__ __launch_bounds__(256) void proj_kernel(const float* __restrict__ hidden,
                                                   const float* __restrict__ Wattn,
                                                   const float* __restrict__ b_attn,
                                                   float* __restrict__ proj) {
  __shared__ float hrow[H_];
  int b = blockIdx.x, tid = threadIdx.x;
  hrow[tid]       = hidden[b * H_ + tid];
  hrow[tid + 256] = hidden[b * H_ + tid + 256];
  __syncthreads();
  float a0 = 0.0f, a1 = 0.0f;
  for (int k = 0; k < H_; ++k) {
    float hv = hrow[k];
    a0 += hv * Wattn[(size_t)k * H_ + tid];
    a1 += hv * Wattn[(size_t)k * H_ + tid + 256];
  }
  proj[b * H_ + tid]       = a0 + b_attn[tid];
  proj[b * H_ + tid + 256] = a1 + b_attn[tid + 256];
}

// ---------------------------------------------------------------------------
// Kernel 3: main WMMA kernel with explicit double-buffered A/B fragments.
// One block = one (b, 16-row s-tile); 8 waves, wave owns 64 N columns.
// ---------------------------------------------------------------------------
__device__ __forceinline__ void load_bfrags(BFrag bfr[4], const uint32_t* bp) {
#pragma unroll
  for (int t = 0; t < 4; ++t) {
    bfr[t].q[0] = *(const uint4*)(bp + t * 256);
    bfr[t].q[1] = *(const uint4*)(bp + t * 256 + 128);
  }
}

__device__ __forceinline__ void load_afrag(BFrag& af, const uint32_t* ap) {
  af.q[0] = *(const uint4*)ap;
  af.q[1] = *(const uint4*)(ap + 4);
}

__device__ __forceinline__ void wmma4(v8f acc[4], const BFrag& af, const BFrag bfr[4]) {
#pragma unroll
  for (int t = 0; t < 4; ++t) {
    acc[t] = __builtin_amdgcn_wmma_f32_16x16x32_bf16(
        false, af.v, false, bfr[t].v, (short)0, acc[t], false, false);
  }
}

__global__ __launch_bounds__(256) void scores_wmma_kernel(
    const float* __restrict__ enc, const int* __restrict__ mask,
    const uint32_t* __restrict__ wpack, const float* __restrict__ proj,
    const float* __restrict__ Wv, float* __restrict__ scores) {
  __shared__ uint32_t A[16 * ASTRIDE];   // 16 rows x 512 bf16-pairs (+pad) = ~33 KB
  __shared__ float red[8][16];
  const int b   = blockIdx.y;
  const int s0  = blockIdx.x * 16;
  const int tid = threadIdx.x;

  // Stage A tile: enc[b, s0:s0+16, 0:1024] fp32 -> packed bf16 pairs in LDS.
  const float4* encp = (const float4*)(enc + ((size_t)b * S_ + s0) * E_);
#pragma unroll
  for (int i = 0; i < 16; ++i) {
    int idx = i * 256 + tid;
    int m   = idx >> 8;    // row 0..15
    int c   = idx & 255;   // float4 column
    float4 v = encp[m * 256 + c];
    uint2 p;
    p.x = packbf2(v.x, v.y);
    p.y = packbf2(v.z, v.w);
    *(uint2*)&A[m * ASTRIDE + c * 2] = p;
  }
  __syncthreads();

  const int lane   = tid & 31;
  const int wave   = tid >> 5;
  const int mrow   = lane & 15;
  const int hiHalf = (lane >> 4) & 1;
  const int akoff  = hiHalf ? 8 : 0;    // A: high half-wave starts at K=16 (8 u32 pairs)

  v8f acc[4] = {};
  const uint32_t* Arow  = &A[mrow * ASTRIDE];
  const uint32_t* wbase = wpack + (size_t)wave * 1024 + lane * 4;  // +kt*8192 per step

  BFrag a0, a1;
  BFrag bb0[4], bb1[4];
  load_afrag(a0, Arow + akoff);
  load_bfrags(bb0, wbase);

  for (int kt = 0; kt < 30; kt += 2) {
    // prefetch kt+1 while computing kt
    load_afrag(a1, Arow + (kt + 1) * 16 + akoff);
    load_bfrags(bb1, wbase + (size_t)(kt + 1) * 8192);
    wmma4(acc, a0, bb0);
    // prefetch kt+2 while computing kt+1
    load_afrag(a0, Arow + (kt + 2) * 16 + akoff);
    load_bfrags(bb0, wbase + (size_t)(kt + 2) * 8192);
    wmma4(acc, a1, bb1);
  }
  // tail: kt = 30, 31
  load_afrag(a1, Arow + 31 * 16 + akoff);
  load_bfrags(bb1, wbase + (size_t)31 * 8192);
  wmma4(acc, a0, bb0);
  wmma4(acc, a1, bb1);

  // Epilogue: D[m][n]: m = r + 8*hiHalf, n = wave*64 + t*16 + mrow
  float part[8];
#pragma unroll
  for (int r = 0; r < 8; ++r) part[r] = 0.0f;
#pragma unroll
  for (int t = 0; t < 4; ++t) {
    int n = wave * 64 + t * 16 + mrow;
    float ph = proj[b * H_ + n];        // proj_h + b_attn pre-summed
    float wv = Wv[n];
#pragma unroll
    for (int r = 0; r < 8; ++r) {
      float e = tanhf(acc[t][r] + ph);
      part[r] += e * wv;
    }
  }
  // butterfly reduce over the 16 lanes of each half-wave
#pragma unroll
  for (int msk = 1; msk <= 8; msk <<= 1) {
#pragma unroll
    for (int r = 0; r < 8; ++r) part[r] += __shfl_xor(part[r], msk, 32);
  }
  if (mrow == 0) {
#pragma unroll
    for (int r = 0; r < 8; ++r) red[wave][hiHalf * 8 + r] = part[r];
  }
  __syncthreads();
  if (tid < 16) {
    float sum = 0.0f;
#pragma unroll
    for (int w = 0; w < 8; ++w) sum += red[w][tid];
    int sidx = s0 + tid;
    int mk = mask[b * S_ + sidx];
    scores[b * S_ + sidx] = (mk == 0) ? NEG_ : sum;
  }
}

// ---------------------------------------------------------------------------
// Kernel 4: per-batch softmax over S, write attn_w, then context GEMV
// streaming enc[b] once (coalesced float4).
// ---------------------------------------------------------------------------
__global__ __launch_bounds__(256) void softmax_ctx_kernel(
    const float* __restrict__ enc, const float* __restrict__ scores,
    float* __restrict__ out) {
  __shared__ float wls[S_];
  __shared__ float red[256];
  int b = blockIdx.x, tid = threadIdx.x;
  float4 sc = ((const float4*)(scores + (size_t)b * S_))[tid];
  float mx = fmaxf(fmaxf(sc.x, sc.y), fmaxf(sc.z, sc.w));
  red[tid] = mx;
  __syncthreads();
  for (int s = 128; s > 0; s >>= 1) {
    if (tid < s) red[tid] = fmaxf(red[tid], red[tid + s]);
    __syncthreads();
  }
  mx = red[0];
  __syncthreads();
  float e0 = __expf(sc.x - mx), e1 = __expf(sc.y - mx);
  float e2 = __expf(sc.z - mx), e3 = __expf(sc.w - mx);
  red[tid] = e0 + e1 + e2 + e3;
  __syncthreads();
  for (int s = 128; s > 0; s >>= 1) {
    if (tid < s) red[tid] += red[tid + s];
    __syncthreads();
  }
  float inv = 1.0f / red[0];
  float4 w4 = make_float4(e0 * inv, e1 * inv, e2 * inv, e3 * inv);
  ((float4*)&wls[0])[tid] = w4;
  ((float4*)(out + (size_t)B_ * E_))[(size_t)b * 256 + tid] = w4;  // attn_w out
  __syncthreads();
  float4 ctx = make_float4(0.0f, 0.0f, 0.0f, 0.0f);
  const float4* ep = (const float4*)(enc + (size_t)b * S_ * E_);
  for (int s = 0; s < S_; ++s) {
    float w = wls[s];
    float4 v = ep[s * 256 + tid];
    ctx.x += w * v.x; ctx.y += w * v.y; ctx.z += w * v.z; ctx.w += w * v.w;
  }
  ((float4*)out)[(size_t)b * 256 + tid] = ctx;                     // context out
}

// ---------------------------------------------------------------------------
extern "C" void kernel_launch(void* const* d_in, const int* in_sizes, int n_in,
                              void* d_out, int out_size, void* d_ws, size_t ws_size,
                              hipStream_t stream) {
  (void)in_sizes; (void)n_in; (void)out_size; (void)ws_size;
  const float* hidden = (const float*)d_in[0];
  const float* enc    = (const float*)d_in[1];
  const int*   mask   = (const int*)d_in[2];
  const float* Wattn  = (const float*)d_in[3];
  const float* b_attn = (const float*)d_in[4];
  const float* Wv     = (const float*)d_in[5];
  float* out = (float*)d_out;

  char* ws = (char*)d_ws;
  uint32_t* wpack = (uint32_t*)ws;                            // 1 MB bf16-packed W_e
  float* proj     = (float*)(ws + (1 << 20));                 // 64 KB  proj_h + b
  float* scores   = (float*)(ws + (1 << 20) + B_ * H_ * 4);   // 128 KB raw scores

  pack_we_kernel<<<1024, 256, 0, stream>>>(Wattn, wpack);
  proj_kernel<<<B_, 256, 0, stream>>>(hidden, Wattn, b_attn, proj);
  scores_wmma_kernel<<<dim3(S_ / 16, B_), 256, 0, stream>>>(enc, mask, wpack, proj, Wv, scores);
  softmax_ctx_kernel<<<B_, 256, 0, stream>>>(enc, scores, out);
}
